// RegionLayerDWv2_70076686402016
// MI455X (gfx1250) — matrix operand order repaired and checked by reading, not compiled, not run on hardware
//
#include <hip/hip_runtime.h>
#include <hip/hip_bf16.h>

// ---------------------------------------------------------------------------
// RegionLayerDWv2 fused kernel for gfx1250 (MI455X).
//
// out = BN2(pw . GELU(BN1(dwconv3x3(x)))) + res . x      (per region r)
//     = (sc2*pw | res) . [Y ; X] + sh2                    (fused K=256 GEMM)
//
// Kernel 1 packs bf16 weights in WMMA A-fragment lane order into d_ws.
// Kernel 2: per (r, b, 4-row strip): stage1 dwconv+BN1+GELU -> LDS bf16
// B-panel [K=256][N=128] (vectorized float4 loads / v4bf stores);
// stage2 bf16 WMMA GEMM 256x256x128 + epilogue.
// ---------------------------------------------------------------------------

#define RNUM   64
#define BATCH  8
#define CIN    128
#define HH     256
#define WW     256
#define COUT   256
#define EPSV   1e-5f

#define KTOT   256      // fused K: 128 (GELU path) + 128 (residual path)
#define NPIX   128      // pixels per workgroup: 4 rows x 32 cols
#define NPAD   136      // LDS B-panel row stride in halves (16B-aligned rows)
#define KSTEPS 8        // KTOT / 32
#define FRAG_HALVES 512 // one 16x32 bf16 A fragment = 32 lanes * 16 halves

typedef __attribute__((ext_vector_type(16))) __bf16 v16bf;
typedef __attribute__((ext_vector_type(8)))  __bf16 v8bf;
typedef __attribute__((ext_vector_type(4)))  __bf16 v4bf;
typedef __attribute__((ext_vector_type(8)))  float  v8f;
typedef __attribute__((ext_vector_type(4)))  float  v4f;

// Load a 16-half (32B) per-lane fragment slice as two 16B vectors.
__device__ __forceinline__ v16bf ld_frag16(const __bf16* p) {
  v8bf lo = *(const v8bf*)(p);
  v8bf hi = *(const v8bf*)(p + 8);
  v16bf r;
#pragma unroll
  for (int i = 0; i < 8; ++i) { r[i] = lo[i]; r[i + 8] = hi[i]; }
  return r;
}

// ---------------------------------------------------------------------------
// Kernel 1a: pack [sc2*pw | res] into bf16 WMMA A-fragment order.
// packA index = (((r*KSTEPS + kt)*16 + mt)*32 + lane)*16 + j
// ISA 16-bit A 16x32 layout: lane -> M = mt*16 + (lane&15);
//   VGPR v=j>>1, half=j&1: v in 0..3 -> k = base8 + j ;
//                          v in 4..7 -> k = 16 + base8 + (j-8)
// ---------------------------------------------------------------------------
__global__ __launch_bounds__(256) void pack_weights_kernel(
    const float* __restrict__ pw_w, const float* __restrict__ res_w,
    const float* __restrict__ g2,  const float* __restrict__ v2,
    __bf16* __restrict__ packA)
{
  size_t idx = (size_t)blockIdx.x * 256 + threadIdx.x;  // 4,194,304 total
  int j    = (int)(idx & 15);
  int lane = (int)((idx >> 4) & 31);
  int mt   = (int)((idx >> 9) & 15);
  int kt   = (int)((idx >> 13) & 7);
  int r    = (int)(idx >> 16);                          // 0..63

  int m = mt * 16 + (lane & 15);
  int base8 = (lane >> 4) * 8;
  int klocal = (j < 8) ? (base8 + j) : (16 + base8 + (j - 8));
  int k = kt * 32 + klocal;

  float val;
  if (k < CIN) {
    float sc2 = g2[r * COUT + m] * rsqrtf(v2[r * COUT + m] + EPSV);
    val = pw_w[((size_t)r * COUT + m) * CIN + k] * sc2;
  } else {
    val = res_w[((size_t)r * COUT + m) * CIN + (k - CIN)];
  }
  packA[idx] = (__bf16)val;
}

// Kernel 1b: sh2[r*COUT+o] = b2 - g2*m2*rsqrt(v2+eps)
__global__ __launch_bounds__(256) void make_sh2_kernel(
    const float* __restrict__ g2, const float* __restrict__ b2,
    const float* __restrict__ m2, const float* __restrict__ v2,
    float* __restrict__ sh2)
{
  int i = blockIdx.x * 256 + threadIdx.x;   // 16384
  float rs = rsqrtf(v2[i] + EPSV);
  sh2[i] = b2[i] - g2[i] * m2[i] * rs;
}

// ---------------------------------------------------------------------------
// Kernel 2: fused dwconv+BN1+GELU -> LDS, then 256x256x128 bf16 WMMA GEMM.
// Grid: 64 regions * 8 batch * 8 strips = 4096 workgroups of 256 threads.
// ---------------------------------------------------------------------------
__global__ __launch_bounds__(256) void region_fused_kernel(
    const float* __restrict__ x,    const float* __restrict__ dw_w,
    const float* __restrict__ g1,   const float* __restrict__ b1,
    const float* __restrict__ m1,   const float* __restrict__ v1,
    const __bf16* __restrict__ packA, const float* __restrict__ sh2,
    float* __restrict__ out)
{
  extern __shared__ __bf16 ldsB[];   // [KTOT][NPAD] bf16 = 69,632 bytes

  const int wg    = blockIdx.x;
  const int strip = wg & 7;          // 4-row strip within tile (0..7)
  const int b     = (wg >> 3) & 7;
  const int r     = wg >> 6;         // region 0..63
  const int gh    = r >> 3, gw = r & 7;
  const int h0    = gh * 32, w0 = gw * 32;
  const int tid   = threadIdx.x;

  // ---- Stage 1: depthwise 3x3 (zero-pad at tile border) + BN1 + exact GELU.
  // 512 tasks = 128 channels x 4 rows; each thread does 2 tasks.
  // Rows are read as aligned float4 blocks (rows 128B-aligned in x), output
  // written as 8-byte v4bf stores into the LDS B-panel.
  const v4f z4 = {};
#pragma unroll
  for (int tt = 0; tt < 2; ++tt) {
    const int task = tid + tt * 256;     // 0..511
    const int c    = task & 127;
    const int row  = task >> 7;          // 0..3
    const int tr   = strip * 4 + row;    // tile row 0..31

    const float* xc = x + (((size_t)b * CIN + c) * HH + (h0 + tr)) * WW + w0;
    const float* pU = xc - WW;
    const float* pD = xc + WW;
    const bool up = (tr > 0), dn = (tr < 31);

    float wk[9];
#pragma unroll
    for (int i = 0; i < 9; ++i) wk[i] = dw_w[((size_t)r * CIN + c) * 9 + i];
    const float rs = rsqrtf(v1[r * CIN + c] + EPSV);
    const float sc = g1[r * CIN + c] * rs;
    const float sh = b1[r * CIN + c] - g1[r * CIN + c] * m1[r * CIN + c] * rs;

    __bf16* yrow = ldsB + c * NPAD + row * 32;          // K = c       (GELU)
    __bf16* xrow = ldsB + (CIN + c) * NPAD + row * 32;  // K = 128 + c (raw)

    // current 4-column block of the three rows, plus left-neighbor scalars
    v4f u = up ? *(const v4f*)(pU) : z4;
    v4f m = *(const v4f*)(xc);
    v4f d = dn ? *(const v4f*)(pD) : z4;
    float lu = 0.f, lm = 0.f, ldn = 0.f;                // column -1 (zero pad)

#pragma unroll
    for (int bk = 0; bk < 8; ++bk) {
      v4f nu, nm, nd;                                   // next block (look-ahead)
      if (bk < 7) {
        const int off = 4 * bk + 4;
        nu = up ? *(const v4f*)(pU + off) : z4;
        nm = *(const v4f*)(xc + off);
        nd = dn ? *(const v4f*)(pD + off) : z4;
      } else {
        nu = z4; nm = z4; nd = z4;                      // column 32 (zero pad)
      }

      v4bf ypack, xpack;
#pragma unroll
      for (int i = 0; i < 4; ++i) {
        const float ul = (i == 0) ? lu  : u[i - 1];
        const float ml = (i == 0) ? lm  : m[i - 1];
        const float dl = (i == 0) ? ldn : d[i - 1];
        const float ur = (i == 3) ? nu[0] : u[i + 1];
        const float mr = (i == 3) ? nm[0] : m[i + 1];
        const float dr = (i == 3) ? nd[0] : d[i + 1];
        float conv = ul   * wk[0] + u[i] * wk[1] + ur * wk[2]
                   + ml   * wk[3] + m[i] * wk[4] + mr * wk[5]
                   + dl   * wk[6] + d[i] * wk[7] + dr * wk[8];
        float vv = conv * sc + sh;
        float gl = 0.5f * vv * (1.f + erff(vv * 0.70710678118654752f)); // exact GELU
        ypack[i] = (__bf16)gl;
        xpack[i] = (__bf16)m[i];
      }
      *(v4bf*)(yrow + 4 * bk) = ypack;   // 8-byte DS store
      *(v4bf*)(xrow + 4 * bk) = xpack;   // 8-byte DS store

      lu = u[3]; lm = m[3]; ldn = d[3];
      u = nu; m = nm; d = nd;
    }
  }
  __syncthreads();

  // ---- Stage 2: D[256 x 128] = A[256 x 256] x B[256 x 128] via WMMA bf16.
  // 8 waves: wave -> (mquad 0..3, nhalf 0..1); wave tile 64(M) x 64(N),
  // 16 f32 16x16 accumulators (128 VGPRs/wave).
  const int lane  = tid & 31;
  const int wave  = tid >> 5;
  const int mquad = wave & 3;
  const int nhalf = wave >> 2;

  const v8f vzero = {};
  v8f acc[4][4];
#pragma unroll
  for (int im = 0; im < 4; ++im)
#pragma unroll
    for (int j = 0; j < 4; ++j) acc[im][j] = vzero;

  const __bf16* Abase = packA + (size_t)r * (KSTEPS * 16 * FRAG_HALVES);

  for (int kt = 0; kt < KSTEPS; ++kt) {
    // B fragments: lane -> N-column (lane&15), 16 halves = K in table order.
    v16bf Bf[4];
    const __bf16* brow = ldsB + (kt * 32 + (lane >> 4) * 8) * NPAD + (lane & 15);
#pragma unroll
    for (int j = 0; j < 4; ++j)
#pragma unroll
      for (int s = 0; s < 16; ++s)
        Bf[j][s] = brow[((s < 8) ? s : (s + 8)) * NPAD + (nhalf * 4 + j) * 16];

#pragma unroll
    for (int im = 0; im < 4; ++im) {
      const int mt = mquad * 4 + im;
      v16bf Af = ld_frag16(Abase + ((size_t)kt * 16 + mt) * FRAG_HALVES + lane * 16);
#pragma unroll
      for (int j = 0; j < 4; ++j)
        acc[im][j] = __builtin_amdgcn_wmma_f32_16x16x32_bf16(
            false, Af, false, Bf[j], (short)0, acc[im][j], false, false);
    }
  }

  // ---- Epilogue: + sh2[o], scatter to NCHW (lanes contiguous along W).
  const int l15 = lane & 15;
  const int hi8 = (lane >> 4) * 8;
#pragma unroll
  for (int im = 0; im < 4; ++im) {
#pragma unroll
    for (int j = 0; j < 4; ++j) {
      const int p = (nhalf * 4 + j) * 16 + l15;    // pixel 0..127
      const int h = h0 + strip * 4 + (p >> 5);
      const int w = w0 + (p & 31);
#pragma unroll
      for (int v = 0; v < 8; ++v) {
        const int o = mquad * 64 + im * 16 + hi8 + v;
        float val = acc[im][j][v] + sh2[r * COUT + o];
        out[(((size_t)b * COUT + o) * HH + h) * WW + w] = val;
      }
    }
  }
}

// ---------------------------------------------------------------------------
// d_in order: x, dw_w, bn1_g, bn1_b, bn1_m, bn1_v, pw_w, bn2_g, bn2_b,
//             bn2_m, bn2_v, res_w
// d_ws layout: [0, 8 MiB) packed bf16 A fragments; [8 MiB, +64 KiB) sh2 f32.
// ---------------------------------------------------------------------------
extern "C" void kernel_launch(void* const* d_in, const int* in_sizes, int n_in,
                              void* d_out, int out_size, void* d_ws, size_t ws_size,
                              hipStream_t stream) {
  const float* x     = (const float*)d_in[0];
  const float* dw_w  = (const float*)d_in[1];
  const float* bn1_g = (const float*)d_in[2];
  const float* bn1_b = (const float*)d_in[3];
  const float* bn1_m = (const float*)d_in[4];
  const float* bn1_v = (const float*)d_in[5];
  const float* pw_w  = (const float*)d_in[6];
  const float* bn2_g = (const float*)d_in[7];
  const float* bn2_b = (const float*)d_in[8];
  const float* bn2_m = (const float*)d_in[9];
  const float* bn2_v = (const float*)d_in[10];
  const float* res_w = (const float*)d_in[11];
  float* out = (float*)d_out;

  __bf16* packA = (__bf16*)d_ws;
  float*  sh2   = (float*)((char*)d_ws + (size_t)8 * 1024 * 1024);

  // 64 r * 8 kt * 16 mt * 512 halves = 4,194,304 halves
  pack_weights_kernel<<<16384, 256, 0, stream>>>(pw_w, res_w, bn2_g, bn2_v, packA);
  make_sh2_kernel<<<64, 256, 0, stream>>>(bn2_g, bn2_b, bn2_m, bn2_v, sh2);

  const size_t smem = (size_t)KTOT * NPAD * sizeof(__bf16);  // 69,632 B
  region_fused_kernel<<<RNUM * BATCH * 8, 256, smem, stream>>>(
      x, dw_w, bn1_g, bn1_b, bn1_m, bn1_v, packA, sh2, out);
}
